// Architecture_11879879540882
// MI455X (gfx1250) — matrix operand order repaired and checked by reading, not compiled, not run on hardware
//
#include <hip/hip_runtime.h>
#include <hip/hip_bf16.h>
#include <math.h>

#define BQ 16
#define SQ 512
#define DQ 256
#define HQ 8
#define DKQ 32
#define MTOK (BQ*SQ)

typedef _Float16 v16h __attribute__((ext_vector_type(16)));
typedef float    v8f  __attribute__((ext_vector_type(8)));

__device__ __forceinline__ float wave_rmax(float v){
#pragma unroll
  for (int m = 16; m > 0; m >>= 1) v = fmaxf(v, __shfl_xor(v, m, 32));
  return v;
}
__device__ __forceinline__ float wave_rsum(float v){
#pragma unroll
  for (int m = 16; m > 0; m >>= 1) v += __shfl_xor(v, m, 32);
  return v;
}

// Pack two contiguous 8-float runs (b128 loads) into the 16-half A/B fragment.
__device__ __forceinline__ v16h pack16(const float* p0, const float* p1){
  const float4 r0 = *(const float4*)(p0);
  const float4 r1 = *(const float4*)(p0 + 4);
  const float4 r2 = *(const float4*)(p1);
  const float4 r3 = *(const float4*)(p1 + 4);
  v16h a;
  a[0]=(_Float16)r0.x; a[1]=(_Float16)r0.y; a[2]=(_Float16)r0.z; a[3]=(_Float16)r0.w;
  a[4]=(_Float16)r1.x; a[5]=(_Float16)r1.y; a[6]=(_Float16)r1.z; a[7]=(_Float16)r1.w;
  a[8]=(_Float16)r2.x; a[9]=(_Float16)r2.y; a[10]=(_Float16)r2.z; a[11]=(_Float16)r2.w;
  a[12]=(_Float16)r3.x; a[13]=(_Float16)r3.y; a[14]=(_Float16)r3.z; a[15]=(_Float16)r3.w;
  return a;
}

// ---------------------------------------------------------------------------
// Full-size WMMA GEMM: C[MTOK,256] = A[MTOK,256] @ Wt[256,256] + bias.
// No bounds checks (all dims exact multiples of tile). 4 waves / block,
// one 16x16 tile per wave, b128 A loads, f16 WMMA with f32 accumulate.
// ---------------------------------------------------------------------------
__global__ __launch_bounds__(128)
void wmma_gemm_full(const float* __restrict__ A,
                    const float* __restrict__ Wt,
                    const float* __restrict__ bias,
                    float* __restrict__ C){
  const int lane = threadIdx.x & 31;
  const int wv   = threadIdx.x >> 5;
  const int n0   = (blockIdx.x * 4 + wv) * 16;
  const int m0   = blockIdx.y * 16;
  const int g    = lane >> 4;
  const int lr   = lane & 15;
  const float* Arow = A + (size_t)(m0 + lr) * DQ;
  const int bc = n0 + lr;

  v8f acc = {0.f,0.f,0.f,0.f,0.f,0.f,0.f,0.f};
#pragma unroll 2
  for (int kt = 0; kt < DQ; kt += 32){
    // A 16x32 fragment: lane holds K run [kt+g*16, +16)
    v16h a = pack16(Arow + kt + g * 16, Arow + kt + g * 16 + 8);
    v16h b;
    const int kb = kt + g * 16;
#pragma unroll
    for (int h = 0; h < 16; ++h)            // B 32x16: strided column reads
      b[h] = (_Float16)Wt[(size_t)(kb + h) * DQ + bc];
    acc = __builtin_amdgcn_wmma_f32_16x16x32_f16(false, a, false, b,
                                                 (short)0, acc, false, false);
  }
  const float bv = bias[bc];
#pragma unroll
  for (int r = 0; r < 8; ++r)
    C[(size_t)(m0 + r + g * 8) * DQ + bc] = acc[r] + bv;
}

// ---------------------------------------------------------------------------
// Guarded generic GEMM (only for tiny launches: M=1 know-query, M=8 lk key).
// ---------------------------------------------------------------------------
__global__ __launch_bounds__(128)
void wmma_gemm_small(const float* __restrict__ A, long lda,
                     const float* __restrict__ Wt,
                     const float* __restrict__ bias,
                     float* __restrict__ C,
                     int M, int N, int K, int act){
  const int lane = threadIdx.x & 31;
  const int wv   = threadIdx.x >> 5;
  const int n0   = (blockIdx.x * 4 + wv) * 16;
  const int m0   = blockIdx.y * 16;
  if (n0 >= N) return;
  const int g  = lane >> 4;
  const int lr = lane & 15;
  const int ar = min(m0 + lr, M - 1);
  const int bc = min(n0 + lr, N - 1);

  v8f acc = {0.f,0.f,0.f,0.f,0.f,0.f,0.f,0.f};
  for (int kt = 0; kt < K; kt += 32){
    v16h a, b;
#pragma unroll
    for (int h = 0; h < 16; ++h){
      int kk = kt + g * 16 + h;
      a[h] = (_Float16)A[(size_t)ar * lda + kk];
    }
    const int kb = kt + g * 16;
#pragma unroll
    for (int h = 0; h < 16; ++h)
      b[h] = (_Float16)Wt[(size_t)(kb + h) * N + bc];
    acc = __builtin_amdgcn_wmma_f32_16x16x32_f16(false, a, false, b,
                                                 (short)0, acc, false, false);
  }
#pragma unroll
  for (int r = 0; r < 8; ++r){
    int m = m0 + r + g * 8;
    int n = n0 + lr;
    if (m < M && n < N){
      float v = acc[r] + bias[n];
      if (act == 1) v = 1.f / (1.f + __expf(-v));
      C[(size_t)m * N + n] = v;
    }
  }
}

// ---------------------------------------------------------------------------
// Fused attention for one (batch, head, 16-row query tile).
// Q,K,V laid out [B,S,H*DK]; q row stride may be 0 (broadcast query).
// Phase 1: WMMA scores -> LDS strip [16][512]
// Phase 2: masked softmax, cumsum distance-decay stats, UNMASKED 2nd softmax
// Phase 3: WMMA probs @ V -> Out [B,S,H*DK]
// ---------------------------------------------------------------------------
#define SPAD 516
__global__ __launch_bounds__(512)
void attention_kernel(const float* __restrict__ Q, long qstride,
                      const float* __restrict__ Kp,
                      const float* __restrict__ Vp,
                      const float* __restrict__ gamma,
                      float* __restrict__ Out,
                      int peek_cur){
  __shared__ float sc[16][SPAD];
  const int mt = blockIdx.x, hh = blockIdx.y, bb = blockIdx.z;
  const int m0   = mt * 16;
  const int lane = threadIdx.x & 31;
  const int wv   = threadIdx.x >> 5;
  const int g    = lane >> 4;
  const int lr   = lane & 15;
  const float scale = 0.17677669529663687f;  // 1/sqrt(32)

  const float* qbase = Q  + (size_t)bb * SQ * (size_t)qstride + hh * DKQ;
  const float* kbase = Kp + ((size_t)bb * SQ) * DQ + hh * DKQ;
  const float* vbase = Vp + ((size_t)bb * SQ) * DQ + hh * DKQ;

  // ---- Phase 1: scores = (Q K^T) * scale -----------------------------------
  v16h afrag;
  {
    const float* qrow = qbase + (size_t)(m0 + lr) * qstride;
    afrag = pack16(qrow + g * 16, qrow + g * 16 + 8);
  }
#pragma unroll
  for (int t = 0; t < 2; ++t){
    const int kt = wv * 2 + t;
    const float* krow = kbase + (size_t)(kt * 16 + lr) * DQ + g * 16;
    v16h b = pack16(krow, krow + 8);   // 16 contiguous K-dim values
    v8f c = {0.f,0.f,0.f,0.f,0.f,0.f,0.f,0.f};
    c = __builtin_amdgcn_wmma_f32_16x16x32_f16(false, afrag, false, b,
                                               (short)0, c, false, false);
#pragma unroll
    for (int r = 0; r < 8; ++r)
      sc[r + g * 8][kt * 16 + lr] = c[r] * scale;
  }
  __syncthreads();

  // ---- Phase 2: per-row statistics (wave w owns row w) ---------------------
  {
    const int row_i = wv;
    const int iglob = m0 + row_i;
    float* row = sc[row_i];
    const float gm   = gamma[hh];
    const float gneg = -log1pf(__expf(gm));        // -softplus(gamma)
    const int limit = peek_cur ? iglob : iglob - 1;
    const int j0 = lane * 16;

    // one-shot b128 LDS read of this lane's 16-score chunk
    float rv[16];
    {
      const float4* p = (const float4*)(row + j0);
#pragma unroll
      for (int q = 0; q < 4; ++q){
        float4 f = p[q];
        rv[q*4+0] = f.x; rv[q*4+1] = f.y; rv[q*4+2] = f.z; rv[q*4+3] = f.w;
      }
    }

    float mx = -3.0e38f;
#pragma unroll
    for (int jj = 0; jj < 16; ++jj)
      if (j0 + jj <= limit) mx = fmaxf(mx, rv[jj]);
    mx = wave_rmax(mx);

    float ev[16];
    float lsum = 0.f;
#pragma unroll
    for (int jj = 0; jj < 16; ++jj){
      float e = (j0 + jj <= limit) ? __expf(rv[jj] - mx) : 0.f;
      ev[jj] = e; lsum += e;
    }
    float tot1 = wave_rsum(lsum);
    float inv1 = (limit >= 0 && tot1 > 0.f) ? (1.f / tot1) : 0.f;

    // prefix scan of per-lane softmax mass
    float lane_tot = lsum * inv1;
    float incl = lane_tot;
#pragma unroll
    for (int d = 1; d < 32; d <<= 1){
      float t = __shfl_up(incl, (unsigned)d, 32);
      if (lane >= d) incl += t;
    }
    float excl = incl - lane_tot;
    float dt = (limit >= 0 && tot1 > 0.f) ? 1.f : 0.f;  // disttot

    float run = excl;
    float tv[16];
    float mx2 = -3.0e38f;
#pragma unroll
    for (int jj = 0; jj < 16; ++jj){
      int j = j0 + jj;
      run += ev[jj] * inv1;                       // distcum
      float pos  = fabsf((float)(iglob - j));
      float dist = sqrtf(fmaxf((dt - run) * pos, 0.f));
      float totl = __expf(dist * gneg);
      totl = fminf(fmaxf(totl, 1e-5f), 1e5f);
      float t2 = rv[jj] * totl;                   // UNMASKED 2nd softmax input
      tv[jj] = t2; mx2 = fmaxf(mx2, t2);
    }
    mx2 = wave_rmax(mx2);
    float s2l = 0.f;
#pragma unroll
    for (int jj = 0; jj < 16; ++jj){
      float e = __expf(tv[jj] - mx2);
      tv[jj] = e; s2l += e;
    }
    float s2   = wave_rsum(s2l);
    float fac  = (1.f / s2) * ((!peek_cur && iglob == 0) ? 0.f : 1.f);
    // b128 LDS write-back of the probabilities
    {
      float4* p = (float4*)(row + j0);
#pragma unroll
      for (int q = 0; q < 4; ++q){
        float4 f;
        f.x = tv[q*4+0]*fac; f.y = tv[q*4+1]*fac;
        f.z = tv[q*4+2]*fac; f.w = tv[q*4+3]*fac;
        p[q] = f;
      }
    }
  }
  __syncthreads();

  // ---- Phase 3: out = probs @ V (waves 0,1 each do one 16-col tile) --------
  if (wv < 2){
    const int nt = wv;
    v8f acc = {0.f,0.f,0.f,0.f,0.f,0.f,0.f,0.f};
    for (int kt = 0; kt < 16; ++kt){
      const int koff = kt * 32;
      const float* prow = &sc[lr][koff + g * 16];
      v16h pa = pack16(prow, prow + 8);           // ds_load_b128 x4
      v16h vb;
#pragma unroll
      for (int h = 0; h < 16; ++h){
        int srow = koff + g * 16 + h;
        vb[h] = (_Float16)vbase[(size_t)srow * DQ + nt * 16 + lr];
      }
      acc = __builtin_amdgcn_wmma_f32_16x16x32_f16(false, pa, false, vb,
                                                   (short)0, acc, false, false);
    }
#pragma unroll
    for (int r = 0; r < 8; ++r){
      int s   = m0 + r + g * 8;
      int col = nt * 16 + lr;
      Out[((size_t)bb * SQ + s) * DQ + hh * DKQ + col] = acc[r];
    }
  }
}

// ---------------------------------------------------------------------------
// y = LN(resid + o) * g + b ; resid row stride may be 0 (broadcast know)
// ---------------------------------------------------------------------------
__global__ __launch_bounds__(256)
void ln_kernel(const float* __restrict__ resid, long rstride,
               const float* __restrict__ o,
               const float* __restrict__ gw, const float* __restrict__ bw,
               float* __restrict__ y){
  const int row = blockIdx.x, t = threadIdx.x;
  __shared__ float red[256];
  __shared__ float mv[2];
  float x = resid[(size_t)row * rstride + t] + o[(size_t)row * DQ + t];
  red[t] = x; __syncthreads();
  for (int s = 128; s > 0; s >>= 1){ if (t < s) red[t] += red[t + s]; __syncthreads(); }
  if (t == 0) mv[0] = red[0] * (1.f / DQ);
  __syncthreads();
  float m = mv[0];
  float d = x - m;
  red[t] = d * d; __syncthreads();
  for (int s = 128; s > 0; s >>= 1){ if (t < s) red[t] += red[t + s]; __syncthreads(); }
  if (t == 0) mv[1] = red[0] * (1.f / DQ);
  __syncthreads();
  y[(size_t)row * DQ + t] = gw[t] * d * rsqrtf(mv[1] + 1e-5f) + bw[t];
}

// ---------------------------------------------------------------------------
// Final readout: beta = key . q_emb, alpha = softmax_h(beta),
// value = sigmoid(h_tok . lv_w + lv_b), out = sum_h alpha_h * value_h
// ---------------------------------------------------------------------------
__global__ __launch_bounds__(256)
void final_kernel(const float* __restrict__ q_emb, const float* __restrict__ hbuf,
                  const float* __restrict__ keyH, const float* __restrict__ lv_w,
                  const float* __restrict__ lv_b, float* __restrict__ out){
  const int n = blockIdx.x;
  const int t = threadIdx.x;
  __shared__ float qv[256], ht[256], red[256], al[8];
  qv[t] = q_emb[(size_t)n * DQ + t];
  ht[t] = hbuf[(size_t)n * DQ + t];
  __syncthreads();
  {
    int h = t >> 5, l = t & 31;
    float p = 0.f;
#pragma unroll
    for (int jj = 0; jj < 8; ++jj){
      int j = l * 8 + jj;
      p += keyH[h * DQ + j] * qv[j];
    }
    red[t] = p; __syncthreads();
    for (int s = 16; s > 0; s >>= 1){ if (l < s) red[t] += red[t + s]; __syncthreads(); }
    if (t == 0){
      float mb = -3.0e38f;
      for (int h2 = 0; h2 < 8; ++h2) mb = fmaxf(mb, red[h2 * 32]);
      float s = 0.f;
      for (int h2 = 0; h2 < 8; ++h2){ float e = __expf(red[h2 * 32] - mb); al[h2] = e; s += e; }
      for (int h2 = 0; h2 < 8; ++h2) al[h2] /= s;
    }
    __syncthreads();
  }
  float wc[32];
#pragma unroll
  for (int k = 0; k < 32; ++k) wc[k] = lv_w[k * DQ + t];
  const float bterm = lv_b[t];
  float acc = 0.f;
#pragma unroll
  for (int h = 0; h < 8; ++h){
    float s = bterm;
#pragma unroll
    for (int k = 0; k < 32; ++k) s += ht[h * 32 + k] * wc[k];
    acc += al[h] * (1.f / (1.f + __expf(-s)));
  }
  out[(size_t)n * DQ + t] = acc;
}

// ---------------------------------------------------------------------------
extern "C" void kernel_launch(void* const* d_in, const int* in_sizes, int n_in,
                              void* d_out, int out_size, void* d_ws, size_t ws_size,
                              hipStream_t stream){
  (void)in_sizes; (void)n_in; (void)out_size; (void)ws_size;
  const float* q_emb  = (const float*)d_in[0];
  const float* qa_emb = (const float*)d_in[1];
  const float* b1_qw  = (const float*)d_in[2];
  const float* b1_qb  = (const float*)d_in[3];
  const float* b1_vw  = (const float*)d_in[4];
  const float* b1_vb  = (const float*)d_in[5];
  const float* b1_ow  = (const float*)d_in[6];
  const float* b1_ob  = (const float*)d_in[7];
  const float* b1_gam = (const float*)d_in[8];
  const float* b1_lng = (const float*)d_in[9];
  const float* b1_lnb = (const float*)d_in[10];
  const float* b2_qw  = (const float*)d_in[11];
  const float* b2_qb  = (const float*)d_in[12];
  const float* b2_vw  = (const float*)d_in[13];
  const float* b2_vb  = (const float*)d_in[14];
  const float* b2_ow  = (const float*)d_in[15];
  const float* b2_ob  = (const float*)d_in[16];
  const float* b2_gam = (const float*)d_in[17];
  const float* b2_lng = (const float*)d_in[18];
  const float* b2_lnb = (const float*)d_in[19];
  const float* b3_qw  = (const float*)d_in[20];
  const float* b3_qb  = (const float*)d_in[21];
  const float* b3_kw  = (const float*)d_in[22];
  const float* b3_kb  = (const float*)d_in[23];
  const float* b3_vw  = (const float*)d_in[24];
  const float* b3_vb  = (const float*)d_in[25];
  const float* b3_ow  = (const float*)d_in[26];
  const float* b3_ob  = (const float*)d_in[27];
  const float* b3_gam = (const float*)d_in[28];
  const float* b3_lng = (const float*)d_in[29];
  const float* b3_lnb = (const float*)d_in[30];
  const float* know   = (const float*)d_in[31];
  const float* lk_w   = (const float*)d_in[32];
  const float* lk_b   = (const float*)d_in[33];
  const float* lv_w   = (const float*)d_in[34];
  const float* lv_b   = (const float*)d_in[35];

  float* Wk = (float*)d_ws;
  const size_t MD = (size_t)MTOK * DQ;     // 2,097,152 floats = 8 MB
  float* bufQK   = Wk;
  float* bufV    = Wk + 1 * MD;
  float* bufAttn = Wk + 2 * MD;
  float* bufO    = Wk + 3 * MD;
  float* hq      = Wk + 4 * MD;
  float* ha      = Wk + 5 * MD;
  float* hh      = Wk + 6 * MD;
  float* q3row   = Wk + 7 * MD;            // 256
  float* keyH    = q3row + 256;            // 8*256

  const dim3 gblk(128);
  const dim3 ggrid(4, MTOK / 16);
  const dim3 agrid(SQ / 16, HQ, BQ);

  // ---- Block 1 (k shares q weights & input -> K == Q) ----
  wmma_gemm_full<<<ggrid, gblk, 0, stream>>>(q_emb, b1_qw, b1_qb, bufQK);
  wmma_gemm_full<<<ggrid, gblk, 0, stream>>>(q_emb, b1_vw, b1_vb, bufV);
  attention_kernel<<<agrid, 512, 0, stream>>>(bufQK, DQ, bufQK, bufV, b1_gam, bufAttn, 1);
  wmma_gemm_full<<<ggrid, gblk, 0, stream>>>(bufAttn, b1_ow, b1_ob, bufO);
  ln_kernel<<<MTOK, 256, 0, stream>>>(q_emb, DQ, bufO, b1_lng, b1_lnb, hq);

  // ---- Block 2 ----
  wmma_gemm_full<<<ggrid, gblk, 0, stream>>>(qa_emb, b2_qw, b2_qb, bufQK);
  wmma_gemm_full<<<ggrid, gblk, 0, stream>>>(qa_emb, b2_vw, b2_vb, bufV);
  attention_kernel<<<agrid, 512, 0, stream>>>(bufQK, DQ, bufQK, bufV, b2_gam, bufAttn, 1);
  wmma_gemm_full<<<ggrid, gblk, 0, stream>>>(bufAttn, b2_ow, b2_ob, bufO);
  ln_kernel<<<MTOK, 256, 0, stream>>>(qa_emb, DQ, bufO, b2_lng, b2_lnb, ha);

  // ---- Block 3 (query = broadcast know; strict causal mask + zero_pad) ----
  wmma_gemm_small<<<dim3(4, 1), gblk, 0, stream>>>(know, DQ, b3_qw, b3_qb, q3row, 1, DQ, DQ, 0);
  wmma_gemm_full<<<ggrid, gblk, 0, stream>>>(hq, b3_kw, b3_kb, bufQK);
  wmma_gemm_full<<<ggrid, gblk, 0, stream>>>(ha, b3_vw, b3_vb, bufV);
  attention_kernel<<<agrid, 512, 0, stream>>>(q3row, 0, bufQK, bufV, b3_gam, bufAttn, 0);
  wmma_gemm_full<<<ggrid, gblk, 0, stream>>>(bufAttn, b3_ow, b3_ob, bufO);
  ln_kernel<<<MTOK, 256, 0, stream>>>(know, 0, bufO, b3_lng, b3_lnb, hh);

  // ---- Readout: keyH[h,:] = sigmoid(know[h*32:+32] @ lk_w + lk_b) ----
  wmma_gemm_small<<<dim3(4, 1), gblk, 0, stream>>>(know, DKQ, lk_w, lk_b, keyH, HQ, DQ, DKQ, 1);
  final_kernel<<<MTOK, 256, 0, stream>>>(q_emb, hh, keyH, lv_w, lv_b, (float*)d_out);
}
